// _MambaMixBlock2D_86904368267766
// MI455X (gfx1250) — compile-verified, matrix operand order checked by reading, not run on hardware
//
#include <hip/hip_runtime.h>
#include <hip/hip_bf16.h>

// ---------------- types -----------------
typedef __attribute__((ext_vector_type(16))) __bf16 v16bf;
typedef __attribute__((ext_vector_type(8)))  __bf16 v8bf;
typedef __attribute__((ext_vector_type(8)))  float  v8f;
typedef __attribute__((ext_vector_type(4)))  float  v4f;

#define DN   256
#define BB   8
#define HH   96
#define WW   96
#define LL   (HH*WW)        // 9216
#define NTOK (BB*LL)        // 73728
#define CCH  96             // scan chunks per sequence
#define TT   (LL/CCH)       // 96 steps per chunk (== HH, so a col-chunk is one column)

__device__ __forceinline__ float sigf(float x){ return 1.f/(1.f+__expf(-x)); }
__device__ __forceinline__ float chan_a(const float* al, int d){
  float a = sigf(al[d]);
  return fminf(fmaxf(a, 1e-4f), 1.f-1e-4f);
}

// ---------------- LayerNorm -> split bf16 hi/lo planes -----------------
__global__ void k_ln(const float* __restrict__ x, const float* __restrict__ g,
                     const float* __restrict__ b, __bf16* __restrict__ hh,
                     __bf16* __restrict__ hl) {
  int tok  = blockIdx.x*8 + (threadIdx.x>>5);   // one wave per token
  int lane = threadIdx.x & 31;
  const float* px = x + (size_t)tok*DN + lane*8;
  float v[8];
  *(v4f*)v     = *(const v4f*)px;
  *(v4f*)(v+4) = *(const v4f*)(px+4);
  float s=0.f, s2=0.f;
  #pragma unroll
  for (int i=0;i<8;i++){ s += v[i]; s2 += v[i]*v[i]; }
  #pragma unroll
  for (int m=16;m>=1;m>>=1){ s += __shfl_xor(s,m,32); s2 += __shfl_xor(s2,m,32); }
  float mu = s*(1.f/DN);
  float var = s2*(1.f/DN) - mu*mu;
  float rs = rsqrtf(var + 1e-5f);
  __bf16 oh[8], ol[8];
  #pragma unroll
  for (int i=0;i<8;i++){
    int c = lane*8+i;
    float hv = (v[i]-mu)*rs*g[c] + b[c];
    __bf16 hi = (__bf16)hv;
    oh[i]=hi; ol[i]=(__bf16)(hv-(float)hi);
  }
  *(v8bf*)(hh + (size_t)tok*DN + lane*8) = *(v8bf*)oh;
  *(v8bf*)(hl + (size_t)tok*DN + lane*8) = *(v8bf*)ol;
}

// -------------- transpose + bf16 split a weight matrix (K x N f32 -> N x K bf16 hi/lo) ------
__global__ void k_tsplit(const float* __restrict__ w, __bf16* __restrict__ th,
                         __bf16* __restrict__ tl, int K, int N) {
  int idx = blockIdx.x*256 + threadIdx.x;       // idx = n*K + k
  if (idx >= K*N) return;
  int n = idx / K, k = idx - n*K;
  float v = w[(size_t)k*N + n];
  __bf16 hi = (__bf16)v;
  th[idx] = hi;
  tl[idx] = (__bf16)(v-(float)hi);
}

// -------------- WMMA fragment load: 16-bit A/B lane layout (ISA 7.12.2) ------
// lane 0-15 : row = tile+lane,    elems 0..7 = K=kb..kb+7,   elems 8..15 = K=kb+16..kb+23
// lane16-31 : row = tile+lane-16, kb is advanced by 8 (handled by caller via ka)
__device__ __forceinline__ v16bf load_frag(const __bf16* __restrict__ base,
                                           int stride, int r, int ka) {
  union { v16bf v; v8bf h[2]; } u;
  const __bf16* p = base + (size_t)r*stride + ka;
  u.h[0] = *(const v8bf*)(p);
  u.h[1] = *(const v8bf*)(p + 16);
  return u.v;
}

// -------------- bf16x3 WMMA 16x64 strip: acc += Ah*Bh + Ah*Bl + Al*Bh -----------
__device__ __forceinline__ void wmma_strip(const __bf16* __restrict__ Ah,
                                           const __bf16* __restrict__ Al,
                                           const __bf16* __restrict__ Bh,
                                           const __bf16* __restrict__ Bl,
                                           int K, int tileM, int tileN,
                                           int lane, v8f acc[4]) {
  const int rowA = tileM + (lane & 15);
  const int klo  = (lane >> 4) * 8;
  for (int kb = 0; kb < K; kb += 32) {
    const int ka = kb + klo;
    if (kb + 32 < K)
      __builtin_prefetch(Ah + (size_t)rowA*K + ka + 32, 0, 3);  // global_prefetch_b8
    v16bf ah = load_frag(Ah, K, rowA, ka);
    v16bf al = load_frag(Al, K, rowA, ka);
    #pragma unroll
    for (int n = 0; n < 4; n++) {
      int colB = tileN + n*16 + (lane & 15);
      v16bf bh = load_frag(Bh, K, colB, ka);
      v16bf bl = load_frag(Bl, K, colB, ka);
      acc[n] = __builtin_amdgcn_wmma_f32_16x16x32_bf16(false, ah, false, bh,
                                                       (short)0, acc[n], false, false);
      acc[n] = __builtin_amdgcn_wmma_f32_16x16x32_bf16(false, ah, false, bl,
                                                       (short)0, acc[n], false, false);
      acc[n] = __builtin_amdgcn_wmma_f32_16x16x32_bf16(false, al, false, bh,
                                                       (short)0, acc[n], false, false);
    }
  }
}

// -------------- GEMM1: h @ w_in + b_in -> x_in | sigmoid(gate) -----------
__global__ void k_gemm_in(const __bf16* __restrict__ Ah, const __bf16* __restrict__ Al,
                          const __bf16* __restrict__ Bh, const __bf16* __restrict__ Bl,
                          const float* __restrict__ bias,
                          float* __restrict__ xin, float* __restrict__ gate) {
  int lane = threadIdx.x & 31, wv = threadIdx.x >> 5;
  int tileM = blockIdx.y*128 + wv*16;
  int tileN = blockIdx.x*64;
  v8f acc[4] = {};
  wmma_strip(Ah, Al, Bh, Bl, DN, tileM, tileN, lane, acc);
  int rbase = tileM + ((lane >> 4) * 8);
  #pragma unroll
  for (int n = 0; n < 4; n++) {
    int col = tileN + n*16 + (lane & 15);
    float bv = bias[col];
    bool isGate = (col >= DN);
    float* dst = isGate ? gate : xin;
    int cc = isGate ? (col - DN) : col;
    #pragma unroll
    for (int r = 0; r < 8; r++) {
      float v = acc[n][r] + bv;
      if (isGate) v = sigf(v);
      dst[(size_t)(rbase + r)*DN + cc] = v;
    }
  }
}

// -------------- GEMM2: (y*gate) @ w_out + b_out + x0 -> out -----------
__global__ void k_gemm_out(const __bf16* __restrict__ Ah, const __bf16* __restrict__ Al,
                           const __bf16* __restrict__ Bh, const __bf16* __restrict__ Bl,
                           const float* __restrict__ bias,
                           const float* __restrict__ x0, float* __restrict__ out) {
  int lane = threadIdx.x & 31, wv = threadIdx.x >> 5;
  int tileM = blockIdx.y*128 + wv*16;
  int tileN = blockIdx.x*64;
  v8f acc[4] = {};
  wmma_strip(Ah, Al, Bh, Bl, DN, tileM, tileN, lane, acc);
  int rbase = tileM + ((lane >> 4) * 8);
  #pragma unroll
  for (int n = 0; n < 4; n++) {
    int col = tileN + n*16 + (lane & 15);
    float bv = bias[col];
    #pragma unroll
    for (int r = 0; r < 8; r++) {
      size_t o = (size_t)(rbase + r)*DN + col;
      out[o] = x0[o] + acc[n][r] + bv;
    }
  }
}

// -------------- depthwise 3x3 conv (SAME) + SiLU -----------
__global__ void k_conv(const float* __restrict__ xin, const float* __restrict__ cw,
                       const float* __restrict__ cb, float* __restrict__ u) {
  int idx = blockIdx.x*256 + threadIdx.x;       // over NTOK * (DN/4)
  int cg  = idx & 63;
  int pix = idx >> 6;
  int w = pix % WW, h = (pix / WW) % HH, b = pix / (WW*HH);
  int c = cg*4;
  v4f acc = {0.f,0.f,0.f,0.f};
  #pragma unroll
  for (int ky=-1; ky<=1; ky++){
    int hy = h+ky; if (hy < 0 || hy >= HH) continue;
    #pragma unroll
    for (int kx=-1; kx<=1; kx++){
      int wx = w+kx; if (wx < 0 || wx >= WW) continue;
      v4f xv = *(const v4f*)(xin + (((size_t)(b*HH+hy)*WW + wx)*DN) + c);
      v4f wv = *(const v4f*)(cw + ((ky+1)*3 + (kx+1))*DN + c);
      acc += xv*wv;
    }
  }
  v4f bv = *(const v4f*)(cb + c);
  float o[4];
  #pragma unroll
  for (int i=0;i<4;i++){ float t = acc[i] + bv[i]; o[i] = t*sigf(t); }
  *(v4f*)(u + (size_t)pix*DN + c) = *(v4f*)o;
}

// -------------- chunked linear-recurrence scans -----------
// Row scan: t = h*W+w.  Writes y (=) fwd, (+=) bwd. Records chunk-end states dirs {0,1}.
__global__ void k_scan_row(const float* __restrict__ u, const float* __restrict__ al,
                           const float* __restrict__ sb, const float* __restrict__ sc,
                           const float* __restrict__ sd, float* __restrict__ y,
                           float* __restrict__ ends) {
  int d = threadIdx.x;
  int j = blockIdx.x % CCH, b = blockIdx.x / CCH;
  float a = chan_a(al,d), bb_=sb[d], cc_=sc[d], dd_=sd[d];
  size_t base = ((size_t)b*LL + (size_t)j*TT)*DN + d;
  float s = 0.f;
  for (int i=0;i<TT;i++){
    float xv = u[base + (size_t)i*DN];
    s = __builtin_fmaf(a, s, bb_*xv);
    y[base + (size_t)i*DN] = 0.25f*(cc_*s + dd_*xv);
  }
  ends[(((size_t)0*BB + b)*CCH + j)*DN + d] = s;
  s = 0.f;
  for (int i=TT-1;i>=0;i--){
    float xv = u[base + (size_t)i*DN];
    s = __builtin_fmaf(a, s, bb_*xv);
    y[base + (size_t)i*DN] += 0.25f*(cc_*s + dd_*xv);
  }
  ends[(((size_t)1*BB + b)*CCH + j)*DN + d] = s;
}

// Col scan: tau = w*H+h; chunk j == column w=j (TT==HH). Accumulates (+=). dirs {2,3}.
__global__ void k_scan_col(const float* __restrict__ u, const float* __restrict__ al,
                           const float* __restrict__ sb, const float* __restrict__ sc,
                           const float* __restrict__ sd, float* __restrict__ y,
                           float* __restrict__ ends) {
  int d = threadIdx.x;
  int j = blockIdx.x % CCH, b = blockIdx.x / CCH;
  float a = chan_a(al,d), bb_=sb[d], cc_=sc[d], dd_=sd[d];
  size_t base = (((size_t)b*HH)*WW + j)*DN + d;   // h=0, w=j
  const size_t st = (size_t)WW*DN;
  float s = 0.f;
  for (int i=0;i<TT;i++){
    float xv = u[base + (size_t)i*st];
    s = __builtin_fmaf(a, s, bb_*xv);
    y[base + (size_t)i*st] += 0.25f*(cc_*s + dd_*xv);
  }
  ends[(((size_t)2*BB + b)*CCH + j)*DN + d] = s;
  s = 0.f;
  for (int i=TT-1;i>=0;i--){
    float xv = u[base + (size_t)i*st];
    s = __builtin_fmaf(a, s, bb_*xv);
    y[base + (size_t)i*st] += 0.25f*(cc_*s + dd_*xv);
  }
  ends[(((size_t)3*BB + b)*CCH + j)*DN + d] = s;
}

// Serial combine of chunk carries: in_{j} = a^T in_{j-1} + end_{j-1}  (reversed for bwd dirs)
__global__ void k_carry(const float* __restrict__ al, const float* __restrict__ ends,
                        float* __restrict__ ins) {
  int tid = blockIdx.x*256 + threadIdx.x;       // 4*BB*DN = 8192
  int d   = tid & 255;
  int b   = (tid >> 8) & 7;
  int dir = tid >> 11;
  float a  = chan_a(al,d);
  float aT = __powf(a, (float)TT);
  size_t base = (((size_t)dir*BB + b)*CCH)*DN + d;
  float in = 0.f;
  if ((dir & 1) == 0) {
    for (int j=0;j<CCH;j++){ ins[base+(size_t)j*DN]=in; in = __builtin_fmaf(aT,in,ends[base+(size_t)j*DN]); }
  } else {
    for (int j=CCH-1;j>=0;j--){ ins[base+(size_t)j*DN]=in; in = __builtin_fmaf(aT,in,ends[base+(size_t)j*DN]); }
  }
}

// y += 0.25*c*(in_f*a^{i+1} + in_b*a^{T-i})   (row chunks, dirs 0/1)
__global__ void k_fix_row(const float* __restrict__ al, const float* __restrict__ sc,
                          const float* __restrict__ ins, float* __restrict__ y) {
  int d = threadIdx.x;
  int j = blockIdx.x % CCH, b = blockIdx.x / CCH;
  float a = chan_a(al,d), cc_ = sc[d];
  float in0 = ins[(((size_t)0*BB+b)*CCH+j)*DN+d];
  float in1 = ins[(((size_t)1*BB+b)*CCH+j)*DN+d];
  if (in0 == 0.f && in1 == 0.f) return;
  float aT = __powf(a,(float)TT);
  float pf = a, pb = aT, ia = 1.f/a, q = 0.25f*cc_;
  size_t base = ((size_t)b*LL + (size_t)j*TT)*DN + d;
  for (int i=0;i<TT;i++){
    y[base+(size_t)i*DN] += q*(in0*pf + in1*pb);
    pf *= a; pb *= ia;
  }
}

__global__ void k_fix_col(const float* __restrict__ al, const float* __restrict__ sc,
                          const float* __restrict__ ins, float* __restrict__ y) {
  int d = threadIdx.x;
  int j = blockIdx.x % CCH, b = blockIdx.x / CCH;
  float a = chan_a(al,d), cc_ = sc[d];
  float in0 = ins[(((size_t)2*BB+b)*CCH+j)*DN+d];
  float in1 = ins[(((size_t)3*BB+b)*CCH+j)*DN+d];
  if (in0 == 0.f && in1 == 0.f) return;
  float aT = __powf(a,(float)TT);
  float pf = a, pb = aT, ia = 1.f/a, q = 0.25f*cc_;
  size_t base = (((size_t)b*HH)*WW + j)*DN + d;
  const size_t st = (size_t)WW*DN;
  for (int i=0;i<TT;i++){
    y[base+(size_t)i*st] += q*(in0*pf + in1*pb);
    pf *= a; pb *= ia;
  }
}

// -------------- g = y * gate -> bf16 hi/lo planes -----------
__global__ void k_pack(const float* __restrict__ y, const float* __restrict__ gate,
                       __bf16* __restrict__ gh, __bf16* __restrict__ gl) {
  size_t idx = (size_t)blockIdx.x*256 + threadIdx.x;
  float v = y[idx]*gate[idx];
  __bf16 hi = (__bf16)v;
  gh[idx] = hi;
  gl[idx] = (__bf16)(v-(float)hi);
}

// =============================== launch ===============================
extern "C" void kernel_launch(void* const* d_in, const int* in_sizes, int n_in,
                              void* d_out, int out_size, void* d_ws, size_t ws_size,
                              hipStream_t stream) {
  const float* x      = (const float*)d_in[0];
  const float* ln_g   = (const float*)d_in[1];
  const float* ln_b   = (const float*)d_in[2];
  const float* w_in   = (const float*)d_in[3];
  const float* b_in   = (const float*)d_in[4];
  const float* conv_w = (const float*)d_in[5];
  const float* conv_b = (const float*)d_in[6];
  const float* a_lg   = (const float*)d_in[7];
  const float* ssm_b  = (const float*)d_in[8];
  const float* ssm_c  = (const float*)d_in[9];
  const float* ssm_d  = (const float*)d_in[10];
  const float* w_out  = (const float*)d_in[11];
  const float* b_out  = (const float*)d_in[12];
  float* out = (float*)d_out;

  // ---- workspace layout (256B aligned regions) ----
  char* ws = (char*)d_ws;
  size_t off = 0;
  auto alloc = [&](size_t bytes)->char* {
    char* p = ws + off;
    off = (off + bytes + 255) & ~(size_t)255;
    return p;
  };
  const size_t nBF = (size_t)NTOK*DN*2, nF = (size_t)NTOK*DN*4;
  __bf16* h_hi  = (__bf16*)alloc(nBF);           // later reused as gy_hi
  __bf16* h_lo  = (__bf16*)alloc(nBF);           // later reused as gy_lo
  float*  xin   = (float*)alloc(nF);             // later reused as y accumulator
  float*  gate  = (float*)alloc(nF);
  float*  u     = (float*)alloc(nF);
  __bf16* wiT_h = (__bf16*)alloc((size_t)2*DN*DN*2);
  __bf16* wiT_l = (__bf16*)alloc((size_t)2*DN*DN*2);
  __bf16* woT_h = (__bf16*)alloc((size_t)DN*DN*2);
  __bf16* woT_l = (__bf16*)alloc((size_t)DN*DN*2);
  float*  ends  = (float*)alloc((size_t)4*BB*CCH*DN*4);
  float*  ins   = (float*)alloc((size_t)4*BB*CCH*DN*4);
  float*  y     = xin;                            // alias: xin dead after conv
  __bf16* gy_hi = h_hi;                           // alias: h planes dead after GEMM1
  __bf16* gy_lo = h_lo;

  // 1) weight transpose + bf16 split (tiny, L2 resident afterwards)
  k_tsplit<<<(2*DN*DN+255)/256, 256, 0, stream>>>(w_in,  wiT_h, wiT_l, DN, 2*DN);
  k_tsplit<<<(DN*DN+255)/256,   256, 0, stream>>>(w_out, woT_h, woT_l, DN, DN);

  // 2) LayerNorm -> bf16 hi/lo
  k_ln<<<NTOK/8, 256, 0, stream>>>(x, ln_g, ln_b, h_hi, h_lo);

  // 3) in_proj GEMM (M=73728, N=512, K=256), fused bias + sigmoid gate split
  k_gemm_in<<<dim3(512/64, NTOK/128), 256, 0, stream>>>(h_hi, h_lo, wiT_h, wiT_l,
                                                        b_in, xin, gate);

  // 4) depthwise 3x3 conv + SiLU
  k_conv<<<(NTOK*(DN/4))/256, 256, 0, stream>>>(xin, conv_w, conv_b, u);

  // 5) chunked bidirectional scans (row then col), carry combine, fixups
  k_scan_row<<<BB*CCH, DN, 0, stream>>>(u, a_lg, ssm_b, ssm_c, ssm_d, y, ends);
  k_scan_col<<<BB*CCH, DN, 0, stream>>>(u, a_lg, ssm_b, ssm_c, ssm_d, y, ends);
  k_carry  <<<(4*BB*DN)/256, 256, 0, stream>>>(a_lg, ends, ins);
  k_fix_row<<<BB*CCH, DN, 0, stream>>>(a_lg, ssm_c, ins, y);
  k_fix_col<<<BB*CCH, DN, 0, stream>>>(a_lg, ssm_c, ins, y);

  // 6) gate multiply -> bf16 hi/lo
  k_pack<<<((size_t)NTOK*DN)/256, 256, 0, stream>>>(y, gate, gy_hi, gy_lo);

  // 7) out_proj GEMM (N=256) + bias + residual
  k_gemm_out<<<dim3(256/64, NTOK/128), 256, 0, stream>>>(gy_hi, gy_lo, woT_h, woT_l,
                                                         b_out, x, out);
  (void)in_sizes; (void)n_in; (void)out_size; (void)ws_size;
}
